// PointGridExtractor_52905407152796
// MI455X (gfx1250) — matrix operand order repaired and checked by reading, not compiled, not run on hardware
//
#include <hip/hip_runtime.h>
#include <stdint.h>

#define NPAD 2048
#define CPAD 512
#define BIGKEY 0x4000000000000000LL // 2^62 sentinel, matches reference

typedef __attribute__((ext_vector_type(16))) __bf16        v16bf;
typedef __attribute__((ext_vector_type(8)))  float         v8f;
typedef __attribute__((ext_vector_type(4)))  unsigned int  v4u;

__device__ __forceinline__ unsigned short f2bf(float x) {
  unsigned u = __float_as_uint(x);
  u += 0x7fffu + ((u >> 16) & 1u);           // round-to-nearest-even
  return (unsigned short)(u >> 16);
}

// ---------------- init ----------------
__global__ void k_init_keys(const long long* __restrict__ coors,
                            long long* __restrict__ keys, int* __restrict__ idx) {
  int i = blockIdx.x * blockDim.x + threadIdx.x;
  if (i >= NPAD) return;
  long long b = coors[4*i], d = coors[4*i+1], h = coors[4*i+2], w = coors[4*i+3];
  keys[i] = ((b * 216 + d) * 7992 + h) * 7056 + w;
  idx[i] = i;
}

__global__ void k_set_int(int* p, int v) { *p = v; }

__global__ void k_init_coords(const long long* __restrict__ coors, const int* __restrict__ idx,
                              int* cD, int* cH, int* cW, int* cB) {
  int i = blockIdx.x * blockDim.x + threadIdx.x;
  if (i >= NPAD) return;
  int s = idx[i];
  cB[i] = (int)coors[4*s]; cD[i] = (int)coors[4*s+1];
  cH[i] = (int)coors[4*s+2]; cW[i] = (int)coors[4*s+3];
}

__global__ void k_init_feats(const float* __restrict__ vf, const int* __restrict__ idx,
                             unsigned short* __restrict__ fb16) {
  int t = blockIdx.x * blockDim.x + threadIdx.x;
  if (t >= NPAD * CPAD) return;
  int i = t >> 9, c = t & 511;
  float v = (c == 0) ? vf[idx[i]] : 0.0f;
  fb16[t] = f2bf(v);
}

// ---------------- bitonic sort of 2048 (key,payload) in LDS ----------------
__global__ __launch_bounds__(1024) void k_sort2048(long long* __restrict__ keys,
                                                   int* __restrict__ vals) {
  __shared__ long long sk[NPAD];
  __shared__ int sv[NPAD];
  int t = threadIdx.x;
  sk[t] = keys[t]; sk[t + 1024] = keys[t + 1024];
  sv[t] = vals[t]; sv[t + 1024] = vals[t + 1024];
  __syncthreads();
  for (int size = 2; size <= NPAD; size <<= 1) {
    for (int stride = size >> 1; stride > 0; stride >>= 1) {
      #pragma unroll
      for (int half = 0; half < 2; ++half) {
        int i = t + half * 1024;
        int j = i ^ stride;
        if (j > i) {
          long long ki = sk[i], kj = sk[j];
          bool up = (i & size) == 0;
          if ((ki > kj) == up) {
            sk[i] = kj; sk[j] = ki;
            int tv = sv[i]; sv[i] = sv[j]; sv[j] = tv;
          }
        }
      }
      __syncthreads();
    }
  }
  keys[t] = sk[t]; keys[t + 1024] = sk[t + 1024];
  vals[t] = sv[t]; vals[t + 1024] = sv[t + 1024];
}

// ---------------- rulebook: per (offset, site) neighbor via binary search ----------------
__global__ void k_rulebook(const long long* __restrict__ skeys, const int* __restrict__ nActP,
                           const int* __restrict__ cD, const int* __restrict__ cH,
                           const int* __restrict__ cW, const int* __restrict__ cB,
                           int* __restrict__ rule, int K, int ksz,
                           long long D, long long H, long long W) {
  int t = blockIdx.x * blockDim.x + threadIdx.x;
  if (t >= K * NPAD) return;
  int o = t / NPAD, i = t - o * NPAD;
  int nAct = *nActP;
  int r = ksz >> 1;
  int dz = o / (ksz * ksz) - r, dy = (o / ksz) % ksz - r, dx = o % ksz - r;
  int res = -1;
  if (i < nAct) {
    long long d = (long long)cD[i] + dz, h = (long long)cH[i] + dy, w = (long long)cW[i] + dx;
    if (d >= 0 && d < D && h >= 0 && h < H && w >= 0 && w < W) {
      long long key = (((long long)cB[i] * D + d) * H + h) * W + w;
      int lo = 0, hi = nAct;
      while (lo < hi) { int mid = (lo + hi) >> 1; if (skeys[mid] < key) lo = mid + 1; else hi = mid; }
      if (lo < nAct && skeys[lo] == key) res = lo;
    }
  }
  rule[o * NPAD + i] = res;
}

// ---------------- weight repack into WMMA B-register layout (bf16) ----------------
// wp[((o*CC+cc)*NT+nt)*512 + lane*16 + e] = W[o][cc*32 + (lane>=16?8:0) + (e<8?e:e+8)][nt*16 + (lane&15)]
__global__ void k_pack_w(const float* __restrict__ W, unsigned short* __restrict__ wp,
                         int K, int Cin, int Cout, int CC, int NT) {
  long long t = (long long)blockIdx.x * blockDim.x + threadIdx.x;
  long long total = (long long)K * CC * NT * 512;
  if (t >= total) return;
  int e = (int)(t & 15);
  int lane = (int)((t >> 4) & 31);
  long long r = t >> 9;
  int nt = (int)(r % NT); r /= NT;
  int cc = (int)(r % CC);
  int o  = (int)(r / CC);
  int k = cc * 32 + ((lane & 16) ? 8 : 0) + (e < 8 ? e : e + 8);
  int n = nt * 16 + (lane & 15);
  float v = (k < Cin) ? W[((long long)o * Cin + k) * Cout + n] : 0.0f;
  wp[t] = f2bf(v);
}

// ---------------- gather-GEMM: one wave -> one 16x32 tile (two N-tiles), bf16 WMMA ----
// Compile-time K/CC/NT: no integer division, fully unrolled cin-chunk loop,
// A fragment reused across two accumulators.
template <int K, int CC, int NT>
__global__ __launch_bounds__(256) void k_conv_wmma(
    const unsigned short* __restrict__ fb16,   // [NPAD][CPAD] bf16, compact-sorted rows
    const unsigned short* __restrict__ wp,     // packed B fragments
    const int* __restrict__ rule,              // [K][NPAD]
    float* __restrict__ outF) {                // [NPAD][CPAD] f32
  constexpr int NTH = NT / 2;                  // nt-pairs per M tile
  constexpr int total = (NPAD / 16) * NTH;
  const int lane = threadIdx.x;                // 0..31 (wave32)
  const int wave = blockIdx.x * 8 + threadIdx.y;
  if (wave >= total) return;
  const int mt = wave / NTH;
  const int nt0 = (wave - mt * NTH) * 2;
  const int m0 = mt * 16;
  const int laneM = lane & 15;
  const int hiHalf = (lane >> 4) & 1;          // lanes 16-31 hold the upper K sub-chunk
  v8f acc0 = {0.f, 0.f, 0.f, 0.f, 0.f, 0.f, 0.f, 0.f};
  v8f acc1 = {0.f, 0.f, 0.f, 0.f, 0.f, 0.f, 0.f, 0.f};

  for (int o = 0; o < K; ++o) {
    int nbr = rule[o * NPAD + m0 + laneM];
    // uniform skip when all 16 rows miss this offset (EXEC stays full for WMMA)
    if (__builtin_amdgcn_ballot_w32(nbr >= 0) == 0u) continue;
    const unsigned short* bbase = wp + ((long long)o * CC * NT + nt0) * 512 + lane * 16;
    if (o + 1 < K)  // speculative prefetch of next offset's B block into L2
      __builtin_prefetch(wp + ((long long)(o + 1) * CC * NT + nt0) * 512 + lane * 16, 0, 1);
    #pragma unroll
    for (int cc = 0; cc < CC; ++cc) {
      v4u alo = {0u, 0u, 0u, 0u}, ahi = {0u, 0u, 0u, 0u};
      if (nbr >= 0) {
        // A layout (16-bit, 16x32): lane<16 -> K {0..7,16..23}; lane>=16 -> K {8..15,24..31}
        const v4u* ap = (const v4u*)(fb16 + (long long)nbr * CPAD + cc * 32 + hiHalf * 8);
        alo = ap[0];       // K0 .. K0+7
        ahi = ap[2];       // K0+16 .. K0+23
      }
      const v4u* bp0 = (const v4u*)(bbase + (long long)cc * NT * 512);
      const v4u* bp1 = (const v4u*)(bbase + (long long)cc * NT * 512 + 512);
      v16bf a  = __builtin_bit_cast(v16bf, __builtin_shufflevector(alo, ahi, 0,1,2,3,4,5,6,7));
      v16bf b0 = __builtin_bit_cast(v16bf, __builtin_shufflevector(bp0[0], bp0[1], 0,1,2,3,4,5,6,7));
      v16bf b1 = __builtin_bit_cast(v16bf, __builtin_shufflevector(bp1[0], bp1[1], 0,1,2,3,4,5,6,7));
      acc0 = __builtin_amdgcn_wmma_f32_16x16x32_bf16(false, a, false, b0, (short)0, acc0,
                                                     false, false);
      acc1 = __builtin_amdgcn_wmma_f32_16x16x32_bf16(false, a, false, b1, (short)0, acc1,
                                                     false, false);
    }
  }
  // C/D layout: VGPR v -> row m0 + v + (hiHalf?8:0), col nt*16 + (lane&15)
  const int col0 = nt0 * 16 + laneM;
  const int rowBase = m0 + hiHalf * 8;
  #pragma unroll
  for (int v = 0; v < 8; ++v) {
    outF[(long long)(rowBase + v) * CPAD + col0]      = acc0[v];
    outF[(long long)(rowBase + v) * CPAD + col0 + 16] = acc1[v];
  }
}

// ---------------- BatchNorm (active rows only) + ReLU ----------------
__global__ __launch_bounds__(256) void k_bn_reduce(const float* __restrict__ x,
                                                   const int* __restrict__ nActP,
                                                   float* __restrict__ mean,
                                                   float* __restrict__ var) {
  int c = blockIdx.x;
  int n = *nActP;
  float s = 0.f, s2 = 0.f;
  for (int i = threadIdx.x; i < n; i += 256) {
    float v = x[(long long)i * CPAD + c];
    s += v; s2 += v * v;
  }
  __shared__ float sh[256], sh2[256];
  sh[threadIdx.x] = s; sh2[threadIdx.x] = s2;
  __syncthreads();
  for (int st = 128; st > 0; st >>= 1) {
    if (threadIdx.x < st) { sh[threadIdx.x] += sh[threadIdx.x + st]; sh2[threadIdx.x] += sh2[threadIdx.x + st]; }
    __syncthreads();
  }
  if (threadIdx.x == 0) {
    float nn = (float)(n > 0 ? n : 1);
    float mu = sh[0] / nn;
    mean[c] = mu;
    var[c] = fmaxf(sh2[0] / nn - mu * mu, 0.f);
  }
}

__global__ void k_bn_apply(const float* __restrict__ x, const float* __restrict__ mean,
                           const float* __restrict__ var, const float* __restrict__ g,
                           const float* __restrict__ bta, const int* __restrict__ nActP, int Cout,
                           float* __restrict__ y, unsigned short* __restrict__ yb) {
  int t = blockIdx.x * blockDim.x + threadIdx.x;
  if (t >= NPAD * CPAD) return;
  int i = t >> 9, c = t & 511;
  float v = 0.f;
  if (i < *nActP && c < Cout) {
    float xx = x[t];
    v = (xx - mean[c]) * rsqrtf(var[c] + 1e-4f) * g[c] + bta[c];
    v = fmaxf(v, 0.f);
  }
  y[t] = v;
  yb[t] = f2bf(v);
}

// ---------------- pooling ----------------
__global__ void k_poolkey(const int* __restrict__ cD, const int* __restrict__ cH,
                          const int* __restrict__ cW, const int* __restrict__ cB,
                          const int* __restrict__ nActP,
                          int sd, int sh, int sw, long long Dn, long long Hn, long long Wn,
                          long long* __restrict__ pk, int* __restrict__ ps) {
  int i = blockIdx.x * blockDim.x + threadIdx.x;
  if (i >= NPAD) return;
  long long k = BIGKEY;
  if (i < *nActP) {
    long long d = cD[i] / sd, h = cH[i] / sh, w = cW[i] / sw;
    k = (((long long)cB[i] * Dn + d) * Hn + h) * Wn + w;
  }
  pk[i] = k; ps[i] = i;
}

__global__ __launch_bounds__(1024) void k_pool_build(
    const long long* __restrict__ pk, int* __restrict__ segOf,
    int* cD, int* cH, int* cW, int* cB,
    long long* __restrict__ skeys, int* __restrict__ nActP,
    long long Dn, long long Hn, long long Wn) {
  __shared__ int head[NPAD];
  __shared__ int cbase[64];
  __shared__ int totalHeads;
  int t = threadIdx.x;
  for (int half = 0; half < 2; ++half) {
    int i = t + half * 1024;
    long long k = pk[i];
    head[i] = (k != BIGKEY) && (i == 0 || pk[i - 1] != k);
  }
  __syncthreads();
  if (t < 64) {
    int s = 0;
    for (int j = 0; j < 32; ++j) s += head[t * 32 + j];
    cbase[t] = s;
  }
  __syncthreads();
  if (t == 0) {
    int run = 0;
    for (int j = 0; j < 64; ++j) { int v = cbase[j]; cbase[j] = run; run += v; }
    totalHeads = run;
    *nActP = run;
  }
  __syncthreads();
  for (int half = 0; half < 2; ++half) {
    int i = t + half * 1024;
    int s = cbase[i >> 5];
    for (int j = i & ~31; j <= i; ++j) s += head[j];
    int seg = s - 1;
    segOf[i] = seg;
    if (head[i]) {
      long long k = pk[i];
      long long w = k % Wn; long long r = k / Wn;
      long long h = r % Hn; r /= Hn;
      long long d = r % Dn; long long b = r / Dn;
      cD[seg] = (int)d; cH[seg] = (int)h; cW[seg] = (int)w; cB[seg] = (int)b;
      skeys[seg] = k;                  // unique sorted keys == next stage's skeys
    }
  }
  __syncthreads();
  for (int i = t; i < NPAD; i += 1024)
    if (i >= totalHeads) skeys[i] = BIGKEY;
}

// post-ReLU values are >= 0, so f32 max == u32 bit-pattern max
__global__ void k_pool_scatter(const long long* __restrict__ pk, const int* __restrict__ ps,
                               const int* __restrict__ segOf,
                               const float* __restrict__ src, float* __restrict__ dst) {
  int t = blockIdx.x * blockDim.x + threadIdx.x;
  if (t >= NPAD * CPAD) return;
  int i = t >> 9, c = t & 511;
  if (pk[i] == BIGKEY) return;
  float v = src[(long long)ps[i] * CPAD + c];
  atomicMax((unsigned int*)(dst + (long long)segOf[i] * CPAD + c), __float_as_uint(v));
}

__global__ void k_tobf16(const float* __restrict__ s, unsigned short* __restrict__ d) {
  int t = blockIdx.x * blockDim.x + threadIdx.x;
  if (t < NPAD * CPAD) d[t] = f2bf(s[t]);
}

__global__ void k_zero_f32(float* __restrict__ p, long long n) {
  long long t = (long long)blockIdx.x * blockDim.x + threadIdx.x;
  if (t < n) p[t] = 0.f;
}

// dense (B, C*D, H, W): out[(((b*512 + c)*2 + d)*111 + h)*98 + w]
__global__ void k_dense(const float* __restrict__ f, const int* __restrict__ cD,
                        const int* __restrict__ cH, const int* __restrict__ cW,
                        const int* __restrict__ cB, const int* __restrict__ nActP,
                        float* __restrict__ out) {
  int t = blockIdx.x * blockDim.x + threadIdx.x;
  if (t >= NPAD * CPAD) return;
  int i = t >> 9, c = t & 511;
  if (i >= *nActP) return;
  long long idx = ((((long long)cB[i] * 512 + c) * 2 + cD[i]) * 111 + cH[i]) * 98 + cW[i];
  out[idx] = f[t];
}

// ---------------- conv dispatch (compile-time configs per layer) ----------------
template <int K, int CC, int NT>
static void launch_conv(const unsigned short* fb16, const unsigned short* wp,
                        const int* rule, float* outF, hipStream_t stream) {
  constexpr int waves = (NPAD / 16) * (NT / 2);
  dim3 blk(32, 8);
  k_conv_wmma<K, CC, NT><<<(waves + 7) / 8, blk, 0, stream>>>(fb16, wp, rule, outF);
}

// ---------------- host orchestration ----------------
extern "C" void kernel_launch(void* const* d_in, const int* in_sizes, int n_in,
                              void* d_out, int out_size, void* d_ws, size_t ws_size,
                              hipStream_t stream) {
  (void)in_sizes; (void)n_in; (void)out_size; (void)ws_size;
  const float*     vf    = (const float*)d_in[0];
  const long long* coors = (const long long*)d_in[1];
  const float* Wt[10]; const float* Gm[10]; const float* Bt[10];
  for (int i = 0; i < 10; ++i) {
    Wt[i] = (const float*)d_in[3 + 3 * i];
    Gm[i] = (const float*)d_in[4 + 3 * i];
    Bt[i] = (const float*)d_in[5 + 3 * i];
  }

  size_t off = 0;
  auto A = [&](size_t b) -> char* {
    char* p = (char*)d_ws + off;
    off = (off + b + 255) & ~(size_t)255;
    return p;
  };
  float*          featsA = (float*)A((size_t)NPAD * CPAD * 4);          // conv out / pooled
  float*          featsB = (float*)A((size_t)NPAD * CPAD * 4);          // BN+ReLU out
  unsigned short* fb16   = (unsigned short*)A((size_t)NPAD * CPAD * 2); // bf16 conv input
  unsigned short* wpack  = (unsigned short*)A((size_t)125 * 256 * 256 * 2);
  int*            rule   = (int*)A((size_t)343 * NPAD * 4);
  long long*      skeys  = (long long*)A((size_t)NPAD * 8);
  int*            sidx   = (int*)A((size_t)NPAD * 4);
  long long*      pkeys  = (long long*)A((size_t)NPAD * 8);
  int*            psrc   = (int*)A((size_t)NPAD * 4);
  int*            segOf  = (int*)A((size_t)NPAD * 4);
  int* cD = (int*)A((size_t)NPAD * 4);
  int* cH = (int*)A((size_t)NPAD * 4);
  int* cW = (int*)A((size_t)NPAD * 4);
  int* cB = (int*)A((size_t)NPAD * 4);
  float* mean = (float*)A(512 * 4);
  float* var  = (float*)A(512 * 4);
  int*   nAct = (int*)A(256);

  static const int KS_[10]   = {7,7,7,7,5,5,3,3,3,3};
  static const int CIN_[10]  = {1,64,64,128,128,256,256,384,384,512};
  static const int COUT_[10] = {64,64,128,128,256,256,384,384,512,512};
  static const long long DIMS_[6][3] = {{216,7992,7056},{72,2664,2352},{24,888,784},
                                        {8,444,392},{4,222,196},{2,111,98}};
  static const int POOLS_[5][3] = {{3,3,3},{3,3,3},{3,2,2},{2,2,2},{2,2,2}};

  const int NC = NPAD * CPAD / 256;  // 4096 blocks for 2048x512 element kernels

  k_init_keys<<<(NPAD + 255) / 256, 256, 0, stream>>>(coors, skeys, sidx);
  k_sort2048<<<1, 1024, 0, stream>>>(skeys, sidx);
  k_set_int<<<1, 1, 0, stream>>>(nAct, NPAD);
  k_init_coords<<<(NPAD + 255) / 256, 256, 0, stream>>>(coors, sidx, cD, cH, cW, cB);
  k_init_feats<<<NC, 256, 0, stream>>>(vf, sidx, fb16);

  for (int s = 0; s < 5; ++s) {
    int ksz = KS_[2 * s];
    int K = ksz * ksz * ksz;
    long long D = DIMS_[s][0], H = DIMS_[s][1], W = DIMS_[s][2];
    k_rulebook<<<(K * NPAD + 255) / 256, 256, 0, stream>>>(skeys, nAct, cD, cH, cW, cB,
                                                           rule, K, ksz, D, H, W);
    for (int l = 2 * s; l < 2 * s + 2; ++l) {
      int Cin = CIN_[l], Cout = COUT_[l];
      int CC = (Cin + 31) / 32, NT = Cout / 16;
      long long pt = (long long)K * CC * NT * 512;
      k_pack_w<<<(unsigned)((pt + 255) / 256), 256, 0, stream>>>(Wt[l], wpack, K, Cin, Cout, CC, NT);
      switch (l) {
        case 0: launch_conv<343, 1,  4>(fb16, wpack, rule, featsA, stream); break;
        case 1: launch_conv<343, 2,  4>(fb16, wpack, rule, featsA, stream); break;
        case 2: launch_conv<343, 2,  8>(fb16, wpack, rule, featsA, stream); break;
        case 3: launch_conv<343, 4,  8>(fb16, wpack, rule, featsA, stream); break;
        case 4: launch_conv<125, 4, 16>(fb16, wpack, rule, featsA, stream); break;
        case 5: launch_conv<125, 8, 16>(fb16, wpack, rule, featsA, stream); break;
        case 6: launch_conv<27,  8, 24>(fb16, wpack, rule, featsA, stream); break;
        case 7: launch_conv<27, 12, 24>(fb16, wpack, rule, featsA, stream); break;
        case 8: launch_conv<27, 12, 32>(fb16, wpack, rule, featsA, stream); break;
        case 9: launch_conv<27, 16, 32>(fb16, wpack, rule, featsA, stream); break;
      }
      k_bn_reduce<<<Cout, 256, 0, stream>>>(featsA, nAct, mean, var);
      k_bn_apply<<<NC, 256, 0, stream>>>(featsA, mean, var, Gm[l], Bt[l], nAct, Cout, featsB, fb16);
    }
    long long Dn = DIMS_[s + 1][0], Hn = DIMS_[s + 1][1], Wn = DIMS_[s + 1][2];
    k_poolkey<<<(NPAD + 255) / 256, 256, 0, stream>>>(cD, cH, cW, cB, nAct,
                                                      POOLS_[s][0], POOLS_[s][1], POOLS_[s][2],
                                                      Dn, Hn, Wn, pkeys, psrc);
    k_sort2048<<<1, 1024, 0, stream>>>(pkeys, psrc);
    k_zero_f32<<<NC, 256, 0, stream>>>(featsA, (long long)NPAD * CPAD);
    k_pool_build<<<1, 1024, 0, stream>>>(pkeys, segOf, cD, cH, cW, cB, skeys, nAct, Dn, Hn, Wn);
    k_pool_scatter<<<NC, 256, 0, stream>>>(pkeys, psrc, segOf, featsB, featsA);
    k_tobf16<<<NC, 256, 0, stream>>>(featsA, fb16);
  }

  long long outN = (long long)2 * 512 * 2 * 111 * 98;
  k_zero_f32<<<(unsigned)((outN + 255) / 256), 256, 0, stream>>>((float*)d_out, outN);
  k_dense<<<NC, 256, 0, stream>>>(featsA, cD, cH, cW, cB, nAct, (float*)d_out);
}
